// LinearAttention_10402410791643
// MI455X (gfx1250) — compile-verified, matrix operand order checked by reading, not run on hardware
//
#include <hip/hip_runtime.h>
#include <hip/hip_bf16.h>

// ---------------------------------------------------------------------------
// Causal linear attention for MI455X (gfx1250, wave32, WMMA).
// Pipeline: cast/transpose -> 3x bf16 WMMA GEMM (Q,K,V) -> chunked linear
// attention scan (WMMA for all 64x64x64 matmuls) -> final WMMA GEMM (f32 out).
// GEMM uses gfx1250 async global->LDS copies (ASYNCcnt) + LDS double buffering.
// ---------------------------------------------------------------------------

typedef __attribute__((ext_vector_type(16))) __bf16 v16bf;
typedef __attribute__((ext_vector_type(8)))  __bf16 v8bf;
typedef __attribute__((ext_vector_type(8)))  float  v8f;

#define B_  2
#define T_  2048
#define D_  1024
#define H_  16
#define HD_ 64
#define BT_ (B_ * T_)      // 4096
#define NCHUNK (T_ / 64)   // 32

// ----- helpers --------------------------------------------------------------

static __device__ __forceinline__ __bf16 f2bf(float f) {
  unsigned u = __builtin_bit_cast(unsigned, f);
  u += 0x7fffu + ((u >> 16) & 1u);             // round-to-nearest-even
  unsigned short s = (unsigned short)(u >> 16);
  return __builtin_bit_cast(__bf16, s);
}

static __device__ __forceinline__ v8f wmma_bf16(v16bf a, v16bf b, v8f c) {
  // D = A(16x32 bf16) * B(32x16 bf16) + C(16x16 f32)
  return __builtin_amdgcn_wmma_f32_16x16x32_bf16(
      /*neg_a=*/false, a, /*neg_b=*/false, b,
      /*c_mod=*/(short)0, c, /*reuse_a=*/false, /*reuse_b=*/false);
}

// LDS byte offset of a generic pointer to __shared__ memory: the generic
// address of an LDS location is {aperture_hi, lds_offset[31:0]} (ISA 10.2),
// so the low 32 bits are the DS address.
static __device__ __forceinline__ unsigned lds_off(const void* p) {
  return (unsigned)(unsigned long long)(uintptr_t)p;
}

// gfx1250 async copy: per-lane 16B global -> LDS, tracked by ASYNCcnt.
static __device__ __forceinline__ void async_ld_b128(unsigned dst_lds,
                                                     const void* src_global) {
  asm volatile("global_load_async_to_lds_b128 %0, %1, off"
               :: "v"(dst_lds),
                  "v"((unsigned long long)(uintptr_t)src_global)
               : "memory");
}

static __device__ __forceinline__ void wait_async0() {
  asm volatile("s_wait_asynccnt 0" ::: "memory");
}

// Row-major fragment load (A operand, or B operand when the source is stored
// transposed so each "column" of B is a contiguous row).
// Per ISA: lanes 0-15 hold row (lane), K = {0..7, 16..23};
//          lanes 16-31 hold row (lane-16), K = {8..15, 24..31}.
static __device__ __forceinline__ v16bf frag_ld(const __bf16* base, int stride, int lane) {
  int row = lane & 15;
  int k0  = (lane & 16) ? 8 : 0;
  const __bf16* p = base + row * stride + k0;
  v8bf lo = *(const v8bf*)(p);       // K = k0 .. k0+7
  v8bf hi = *(const v8bf*)(p + 16);  // K = k0+16 .. k0+23
  return __builtin_shufflevector(lo, hi, 0,1,2,3,4,5,6,7,8,9,10,11,12,13,14,15);
}

// Fragment load where the logical matrix is the TRANSPOSE of what is stored:
// element (row=n, k) = stored[k][n]  (strided gather, used for K^T as A-op).
static __device__ __forceinline__ v16bf frag_ld_t(const __bf16* base, int stride, int lane) {
  int n  = lane & 15;
  int k0 = (lane & 16) ? 8 : 0;
  v16bf a;
#pragma unroll
  for (int i = 0; i < 8; ++i) a[i]     = base[(k0 + i) * stride + n];
#pragma unroll
  for (int i = 0; i < 8; ++i) a[i + 8] = base[(k0 + 16 + i) * stride + n];
  return a;
}

// ----- elementwise casts ----------------------------------------------------

__global__ __launch_bounds__(256)
void cast_f32_to_bf16(const float* __restrict__ src, __bf16* __restrict__ dst, int n) {
  for (int i = blockIdx.x * blockDim.x + threadIdx.x; i < n;
       i += gridDim.x * blockDim.x)
    dst[i] = f2bf(src[i]);
}

// Wt[k][n] = (bf16) W[n][k]   (so GEMM computes x @ W^T with K-major operand)
__global__ __launch_bounds__(256)
void transpose_cast_w(const float* __restrict__ W, __bf16* __restrict__ Wt, int D) {
  __shared__ float tile[32][33];
  const int bx = blockIdx.x * 32;   // input row (n)
  const int by = blockIdx.y * 32;   // input col (k)
  const int tx = threadIdx.x & 31;
  const int ty = threadIdx.x >> 5;  // 0..7
#pragma unroll
  for (int i = ty; i < 32; i += 8)
    tile[i][tx] = W[(size_t)(bx + i) * D + by + tx];
  __syncthreads();
#pragma unroll
  for (int i = ty; i < 32; i += 8)
    Wt[(size_t)(by + i) * D + bx + tx] = f2bf(tile[tx][i]);
}

// ----- WMMA GEMM: C[M][N] = A[M][K](bf16) * B[K][N](bf16) + bias ------------
// Block tile 128x128, 256 threads = 8 waves (4x2), wave tile 32x64 (2x4 WMMA).
// Double-buffered LDS; A tiles staged with async global->LDS copies issued
// before the WMMA work of the current iteration (latency hidden by matrix ops).

__global__ __launch_bounds__(256)
void gemm_bf16_wmma(const __bf16* __restrict__ A, const __bf16* __restrict__ B,
                    const float* __restrict__ bias,
                    float* __restrict__ Cf, __bf16* __restrict__ Cb,
                    int M, int N, int K, int out_f32) {
  __shared__ __align__(16) __bf16 As[2][128][32];  // [buf][m][k]
  __shared__ __align__(16) __bf16 Bt[2][128][32];  // [buf][n][k] (transposed)

  const int tid  = threadIdx.x;
  const int lane = tid & 31;
  const int wave = tid >> 5;
  const int wm   = wave >> 1;               // 0..3
  const int wn   = wave & 1;                // 0..1
  const int m0   = blockIdx.y * 128;
  const int n0   = blockIdx.x * 128;

  v8f acc[2][4];
#pragma unroll
  for (int i = 0; i < 2; ++i)
#pragma unroll
    for (int j = 0; j < 4; ++j) acc[i][j] = {};

  const int a_row = tid >> 1;               // 0..127
  const int a_kc  = (tid & 1) * 16;         // 0 or 16
  const int b_k   = tid >> 3;               // 0..31
  const int b_nc  = (tid & 7) * 16;         // 0..112

  const __bf16* a_src0 = A + (size_t)(m0 + a_row) * K + a_kc;
  const __bf16* b_src0 = B + (size_t)b_k * N + n0 + b_nc;

  // prologue: stage k0 = 0 into buffer 0
  async_ld_b128(lds_off(&As[0][a_row][a_kc]),     a_src0);
  async_ld_b128(lds_off(&As[0][a_row][a_kc + 8]), a_src0 + 8);
  {
    v8bf b0 = *(const v8bf*)(b_src0);
    v8bf b1 = *(const v8bf*)(b_src0 + 8);
#pragma unroll
    for (int i = 0; i < 8; ++i) Bt[0][b_nc + i][b_k]     = b0[i];
#pragma unroll
    for (int i = 0; i < 8; ++i) Bt[0][b_nc + 8 + i][b_k] = b1[i];
  }
  wait_async0();
  __syncthreads();

  int cur = 0;
  for (int k0 = 0; k0 < K; k0 += 32) {
    const int  nxt       = cur ^ 1;
    const bool have_next = (k0 + 32) < K;

    v8bf nb0{}, nb1{};
    if (have_next) {
      // issue async A copy for k0+32 into the idle buffer (overlaps compute)
      const __bf16* asrc = a_src0 + k0 + 32;
      async_ld_b128(lds_off(&As[nxt][a_row][a_kc]),     asrc);
      async_ld_b128(lds_off(&As[nxt][a_row][a_kc + 8]), asrc + 8);
      // issue global loads for next B tile into registers
      const __bf16* bsrc = b_src0 + (size_t)(k0 + 32) * N;
      nb0 = *(const v8bf*)(bsrc);
      nb1 = *(const v8bf*)(bsrc + 8);
    }

    // compute from current buffer
    v16bf af[2], bf[4];
#pragma unroll
    for (int mt = 0; mt < 2; ++mt)
      af[mt] = frag_ld(&As[cur][wm * 32 + mt * 16][0], 32, lane);
#pragma unroll
    for (int nt = 0; nt < 4; ++nt)
      bf[nt] = frag_ld(&Bt[cur][wn * 64 + nt * 16][0], 32, lane);
#pragma unroll
    for (int mt = 0; mt < 2; ++mt)
#pragma unroll
      for (int nt = 0; nt < 4; ++nt)
        acc[mt][nt] = wmma_bf16(af[mt], bf[nt], acc[mt][nt]);

    if (have_next) {
      // transpose-scatter next B tile into the idle buffer
#pragma unroll
      for (int i = 0; i < 8; ++i) Bt[nxt][b_nc + i][b_k]     = nb0[i];
#pragma unroll
      for (int i = 0; i < 8; ++i) Bt[nxt][b_nc + 8 + i][b_k] = nb1[i];
      wait_async0();
    }
    __syncthreads();
    cur = nxt;
  }

  // epilogue: C layout per ISA (VGPR r -> M=r / r+8; lane%16 -> N)
#pragma unroll
  for (int mt = 0; mt < 2; ++mt) {
#pragma unroll
    for (int nt = 0; nt < 4; ++nt) {
      const int rowb = m0 + wm * 32 + mt * 16 + ((lane & 16) >> 1);
      const int col  = n0 + wn * 64 + nt * 16 + (lane & 15);
      const float bb = bias[col];
#pragma unroll
      for (int r = 0; r < 8; ++r) {
        const float v = acc[mt][nt][r] + bb;
        const size_t idx = (size_t)(rowb + r) * N + col;
        if (out_f32) Cf[idx] = v;
        else         Cb[idx] = f2bf(v);
      }
    }
  }
}

// ----- chunked causal linear attention (one block per (b,h)) ----------------
// State S[n][p] (64x64 f32) carried across chunks; per chunk of 64 tokens:
//   A = QK^T (masked, intra-chunk), Y = (A.mask)@V + Q@S, S += K^T@V.

__global__ __launch_bounds__(128)
void linattn_wmma(const __bf16* __restrict__ Qg, const __bf16* __restrict__ Kg,
                  const __bf16* __restrict__ Vg, __bf16* __restrict__ Yg) {
  __shared__ __align__(16) __bf16 Qs[64][64];    // [t][d]
  __shared__ __align__(16) __bf16 Ks[64][64];    // [s][d]
  __shared__ __align__(16) __bf16 Vt[64][64];    // [p][s]  (V transposed)
  __shared__ __align__(16) __bf16 Am[64][64];    // masked QK^T, [t][s]
  __shared__ __align__(16) float  Sf[64][64];    // state f32 [n][p]
  __shared__ __align__(16) __bf16 Sbt[64][64];   // state bf16, transposed [p][n]

  const int tid  = threadIdx.x;
  const int lane = tid & 31;
  const int wave = tid >> 5;                 // 0..3
  const int bh   = blockIdx.x;               // 0..31
  const int b    = bh >> 4;
  const int h    = bh & 15;
  const size_t base = (size_t)b * T_ * D_ + (size_t)h * HD_;

  for (int i = tid; i < 64 * 64; i += 128) {
    Sf[i >> 6][i & 63]  = 0.0f;
    Sbt[i >> 6][i & 63] = f2bf(0.0f);
  }
  __syncthreads();

  const int ld_row  = tid >> 1;              // 0..63
  const int ld_half = (tid & 1) * 32;        // 0 or 32

  for (int c = 0; c < NCHUNK; ++c) {
    // ---- stage Q, K via async global->LDS; V transposed through registers --
    {
      const size_t rbase = base + (size_t)(c * 64 + ld_row) * D_ + ld_half;
      const __bf16* qsrc = Qg + rbase;
      const __bf16* ksrc = Kg + rbase;
      const __bf16* vsrc = Vg + rbase;
      async_ld_b128(lds_off(&Qs[ld_row][ld_half]),     qsrc);
      async_ld_b128(lds_off(&Qs[ld_row][ld_half + 8]), qsrc + 8);
      async_ld_b128(lds_off(&Qs[ld_row][ld_half + 16]), qsrc + 16);
      async_ld_b128(lds_off(&Qs[ld_row][ld_half + 24]), qsrc + 24);
      async_ld_b128(lds_off(&Ks[ld_row][ld_half]),     ksrc);
      async_ld_b128(lds_off(&Ks[ld_row][ld_half + 8]), ksrc + 8);
      async_ld_b128(lds_off(&Ks[ld_row][ld_half + 16]), ksrc + 16);
      async_ld_b128(lds_off(&Ks[ld_row][ld_half + 24]), ksrc + 24);
      v8bf vv[4];
#pragma unroll
      for (int j = 0; j < 4; ++j) vv[j] = *(const v8bf*)(vsrc + j * 8);
#pragma unroll
      for (int j = 0; j < 4; ++j)
#pragma unroll
        for (int i = 0; i < 8; ++i)
          Vt[ld_half + j * 8 + i][ld_row] = vv[j][i];
      if (c + 1 < NCHUNK)  // speculative prefetch of next chunk (L2 warm)
        __builtin_prefetch(vsrc + (size_t)64 * D_, 0, 1);
    }
    wait_async0();
    __syncthreads();

    const int tb = wave * 16;                 // this wave's 16 rows

    // ---- phase 1: A = Q K^T, causal mask, stage to LDS as bf16 ----
    {
      v16bf qf0 = frag_ld(&Qs[tb][0], 64, lane);
      v16bf qf1 = frag_ld(&Qs[tb][32], 64, lane);
#pragma unroll
      for (int nt = 0; nt < 4; ++nt) {
        v8f a = {};
        a = wmma_bf16(qf0, frag_ld(&Ks[nt * 16][0], 64, lane), a);
        a = wmma_bf16(qf1, frag_ld(&Ks[nt * 16][32], 64, lane), a);
        const int trow = tb + ((lane & 16) >> 1);
        const int s    = nt * 16 + (lane & 15);
#pragma unroll
        for (int r = 0; r < 8; ++r) {
          const float v = (s <= trow + r) ? a[r] : 0.0f;
          Am[trow + r][s] = f2bf(v);
        }
      }
    }
    __syncthreads();

    // ---- phase 2: Y = Am @ V + Q @ S  (inter-chunk uses bf16 state copy) ----
    {
      v16bf am0 = frag_ld(&Am[tb][0], 64, lane);
      v16bf am1 = frag_ld(&Am[tb][32], 64, lane);
      v16bf qf0 = frag_ld(&Qs[tb][0], 64, lane);
      v16bf qf1 = frag_ld(&Qs[tb][32], 64, lane);
#pragma unroll
      for (int pt = 0; pt < 4; ++pt) {
        v8f y = {};
        y = wmma_bf16(am0, frag_ld(&Vt[pt * 16][0], 64, lane), y);
        y = wmma_bf16(am1, frag_ld(&Vt[pt * 16][32], 64, lane), y);
        y = wmma_bf16(qf0, frag_ld(&Sbt[pt * 16][0], 64, lane), y);
        y = wmma_bf16(qf1, frag_ld(&Sbt[pt * 16][32], 64, lane), y);
        const int trow = tb + ((lane & 16) >> 1);
        const int p    = pt * 16 + (lane & 15);
#pragma unroll
        for (int r = 0; r < 8; ++r) {
          const size_t idx = base + (size_t)(c * 64 + trow + r) * D_ + p;
          Yg[idx] = f2bf(y[r]);
        }
      }
    }
    __syncthreads();  // all reads of Sbt/Vt for Y done

    // ---- phase 3: S += K^T @ V (f32 state in LDS; refresh bf16 copy) ----
    {
      // A-op rows are n = tb..tb+15: element (n, s) = Ks[s][n]
      v16bf kf0 = frag_ld_t(&Ks[0][tb], 64, lane);   // K = s in 0..31
      v16bf kf1 = frag_ld_t(&Ks[32][tb], 64, lane);  // K = s in 32..63
#pragma unroll
      for (int pt = 0; pt < 4; ++pt) {
        const int nrow = tb + ((lane & 16) >> 1);
        const int p    = pt * 16 + (lane & 15);
        v8f cacc;
#pragma unroll
        for (int r = 0; r < 8; ++r) cacc[r] = Sf[nrow + r][p];
        cacc = wmma_bf16(kf0, frag_ld(&Vt[pt * 16][0], 64, lane), cacc);
        cacc = wmma_bf16(kf1, frag_ld(&Vt[pt * 16][32], 64, lane), cacc);
#pragma unroll
        for (int r = 0; r < 8; ++r) {
          Sf[nrow + r][p]  = cacc[r];
          Sbt[p][nrow + r] = f2bf(cacc[r]);
        }
      }
    }
    __syncthreads();  // state updated before next chunk overwrites tiles
  }
}

// ----- launcher -------------------------------------------------------------

extern "C" void kernel_launch(void* const* d_in, const int* in_sizes, int n_in,
                              void* d_out, int out_size, void* d_ws, size_t ws_size,
                              hipStream_t stream) {
  const float* x  = (const float*)d_in[0];
  const float* Wq = (const float*)d_in[1];
  const float* bq = (const float*)d_in[2];
  const float* Wk = (const float*)d_in[3];
  const float* bk = (const float*)d_in[4];
  const float* Wv = (const float*)d_in[5];
  const float* bv = (const float*)d_in[6];
  const float* Wo = (const float*)d_in[7];
  const float* bo = (const float*)d_in[8];
  float* out = (float*)d_out;

  char* ws = (char*)d_ws;
  const size_t MB = 1u << 20;
  __bf16* xb  = (__bf16*)(ws + 0 * MB);    // 8 MB  [BT][D]
  __bf16* wqT = (__bf16*)(ws + 8 * MB);    // 2 MB  [K][N]
  __bf16* wkT = (__bf16*)(ws + 10 * MB);
  __bf16* wvT = (__bf16*)(ws + 12 * MB);
  __bf16* woT = (__bf16*)(ws + 14 * MB);
  __bf16* Qb  = (__bf16*)(ws + 16 * MB);   // 8 MB each
  __bf16* Kb  = (__bf16*)(ws + 24 * MB);
  __bf16* Vb  = (__bf16*)(ws + 32 * MB);
  __bf16* Yb  = (__bf16*)(ws + 40 * MB);   // end: 48 MB

  const int n_x = BT_ * D_;
  cast_f32_to_bf16<<<dim3(4096), dim3(256), 0, stream>>>(x, xb, n_x);

  dim3 tgrid(D_ / 32, D_ / 32);
  transpose_cast_w<<<tgrid, dim3(256), 0, stream>>>(Wq, wqT, D_);
  transpose_cast_w<<<tgrid, dim3(256), 0, stream>>>(Wk, wkT, D_);
  transpose_cast_w<<<tgrid, dim3(256), 0, stream>>>(Wv, wvT, D_);
  transpose_cast_w<<<tgrid, dim3(256), 0, stream>>>(Wo, woT, D_);

  dim3 ggrid(D_ / 128, BT_ / 128);  // (8, 32)
  gemm_bf16_wmma<<<ggrid, dim3(256), 0, stream>>>(xb, wqT, bq, nullptr, Qb,
                                                  BT_, D_, D_, 0);
  gemm_bf16_wmma<<<ggrid, dim3(256), 0, stream>>>(xb, wkT, bk, nullptr, Kb,
                                                  BT_, D_, D_, 0);
  gemm_bf16_wmma<<<ggrid, dim3(256), 0, stream>>>(xb, wvT, bv, nullptr, Vb,
                                                  BT_, D_, D_, 0);

  linattn_wmma<<<dim3(B_ * H_), dim3(128), 0, stream>>>(Qb, Kb, Vb, Yb);

  gemm_bf16_wmma<<<ggrid, dim3(256), 0, stream>>>(Yb, woT, bo, out, nullptr,
                                                  BT_, D_, D_, 1);
}
